// RelativePositionEmbedding_48301202211374
// MI455X (gfx1250) — matrix operand order repaired and checked by reading, not compiled, not run on hardware
//
#include <hip/hip_runtime.h>

typedef __attribute__((ext_vector_type(16))) _Float16 v16h;
typedef __attribute__((ext_vector_type(8)))  _Float16 v8h;
typedef __attribute__((ext_vector_type(4)))  _Float16 v4h;
typedef __attribute__((ext_vector_type(8)))  float    v8f;
typedef __attribute__((ext_vector_type(4)))  float    v4f;

#define D_MODEL 256
#define NHEADS  4
#define HD      64
#define SEQ     1024
#define BATCH   8
#define MAXDIST 7
#define TBL     ((2*MAXDIST+1)*(2*MAXDIST+1))   // 225

static __device__ __forceinline__ v8f wmma_f16(const v16h& a, const v16h& b, const v8f& c) {
  // D = A(16x32 f16) * B(32x16 f16) + C(16x16 f32)
  return __builtin_amdgcn_wmma_f32_16x16x32_f16(false, a, false, b, (short)0, c, false, false);
}

static __device__ __forceinline__ v16h cat8(v8h lo, v8h hi) {
  return __builtin_shufflevector(lo, hi, 0, 1, 2, 3, 4, 5, 6, 7,
                                         8, 9, 10, 11, 12, 13, 14, 15);
}

static __device__ __forceinline__ int clampi(int v, int lo, int hi) {
  return v < lo ? lo : (v > hi ? hi : v);
}

// ---------------------------------------------------------------------------
// Kernel 1: Q = (x @ Wq^T + bq) * 0.125 ; K = x @ Wk^T + bk  (grid.z selects)
// f16 outputs (halved intermediate traffic), f32 WMMA accumulation.
// Tiles staged K-contiguous so every WMMA fragment = two ds_load_b128.
// ---------------------------------------------------------------------------
__global__ __launch_bounds__(256)
void gemm_proj_kernel(const float* __restrict__ X,
                      const float* __restrict__ Wq, const float* __restrict__ bq,
                      const float* __restrict__ Wk, const float* __restrict__ bk,
                      _Float16* __restrict__ Qbuf, _Float16* __restrict__ Kbuf)
{
  __shared__ _Float16 As [64][40];   // A rows   x 32 k  (stride 80B, mult of 16)
  __shared__ _Float16 BsT[64][40];   // B^T cols x 32 k

  const float* Wm;  const float* bias;  _Float16* Out;
  if (blockIdx.z == 0) { Wm = Wq; bias = bq; Out = Qbuf; }
  else                 { Wm = Wk; bias = bk; Out = Kbuf; }
  const float osc = (blockIdx.z == 0) ? 0.125f : 1.0f;   // fold 1/sqrt(hd) into Q

  const int m0   = blockIdx.x * 64;
  const int n0   = blockIdx.y * 64;
  const int tid  = threadIdx.x;
  const int lane = tid & 31;
  const int wid  = tid >> 5;
  const int wm   = wid & 3;
  const int wn   = wid >> 2;
  const int nn   = lane & 15;
  const int half = lane >> 4;

  v8f acc0 = {}; v8f acc1 = {};

  for (int kk = 0; kk < D_MODEL; kk += 32) {
    #pragma unroll
    for (int i = 0; i < 2; ++i) {
      const int g = tid + i * 256;            // 0..511 quad-groups
      const int r  = g >> 3;                  // row/col 0..63
      const int k4 = (g & 7) << 2;            // k offset 0..28
      v4f fa = *(const v4f*)(X  + (size_t)(m0 + r) * D_MODEL + kk + k4);
      *(v4h*)&As[r][k4]  = __builtin_convertvector(fa, v4h);
      v4f fb = *(const v4f*)(Wm + (size_t)(n0 + r) * D_MODEL + kk + k4);
      *(v4h*)&BsT[r][k4] = __builtin_convertvector(fb, v4h);
    }
    if (kk + 32 < D_MODEL) {
      __builtin_prefetch(X + (size_t)(m0 + (tid & 63)) * D_MODEL + kk + 32, 0, 1);
    }
    __syncthreads();

    const int arow = wm * 16 + nn;
    v16h a  = cat8(*(const v8h*)&As[arow][half * 16],
                   *(const v8h*)&As[arow][half * 16 + 8]);
    const int bcol = wn * 32 + nn;
    v16h b0 = cat8(*(const v8h*)&BsT[bcol][half * 16],
                   *(const v8h*)&BsT[bcol][half * 16 + 8]);
    v16h b1 = cat8(*(const v8h*)&BsT[bcol + 16][half * 16],
                   *(const v8h*)&BsT[bcol + 16][half * 16 + 8]);
    acc0 = wmma_f16(a, b0, acc0);
    acc1 = wmma_f16(a, b1, acc1);
    __syncthreads();
  }

  #pragma unroll
  for (int r = 0; r < 8; ++r) {
    const int row = m0 + wm * 16 + r + half * 8;   // C layout: M = r + half*8
    const int c0  = n0 + wn * 32 + nn;             // C layout: N = lane%16
    Out[(size_t)row * D_MODEL + c0]      = (_Float16)((acc0[r] + bias[c0]) * osc);
    Out[(size_t)row * D_MODEL + c0 + 16] = (_Float16)((acc1[r] + bias[c0 + 16]) * osc);
  }
}

// ---------------------------------------------------------------------------
// Kernel 2: fused flash attention + relative-position bias.
// Block = (b, head, 128 queries); 8 waves x 16 queries.
// S = K_tile(A) x Q(B) -> D[M=key, N=query]; softmax axis = M handled with
// per-lane VGPR reduce + shfl_xor(16) partner combine.
// ---------------------------------------------------------------------------
__global__ __launch_bounds__(256)
void attention_kernel(const _Float16* __restrict__ Qbuf,
                      const _Float16* __restrict__ Kbuf,
                      const float* __restrict__ X,
                      const float* __restrict__ bias_table,
                      _Float16* __restrict__ Obuf)
{
  __shared__ float    bt[TBL * NHEADS];   // 3.6 KB bias table
  __shared__ _Float16 Ks [32][72];        // keys x hd   (stride 144B, mult 16)
  __shared__ _Float16 VsT[64][40];        // hd   x keys (stride 80B,  mult 16)

  const int tid  = threadIdx.x;
  const int lane = tid & 31;
  const int wid  = tid >> 5;
  const int nn   = lane & 15;
  const int half = lane >> 4;
  const int h    = blockIdx.y;
  const int b    = blockIdx.z;

  for (int i = tid; i < TBL * NHEADS; i += 256) bt[i] = bias_table[i];

  const int qg = blockIdx.x * 128 + wid * 16 + nn;   // lane's query column
  const int hq = qg >> 5, wq = qg & 31;

  // Q fragments (B operand, already scaled by 1/8 in kernel 1): 4x b128 loads
  const _Float16* qrow = Qbuf + ((size_t)(b * SEQ + qg)) * D_MODEL + h * HD;
  const v16h qf0 = cat8(*(const v8h*)(qrow + half * 8),
                        *(const v8h*)(qrow + 16 + half * 8));
  const v16h qf1 = cat8(*(const v8h*)(qrow + 32 + half * 8),
                        *(const v8h*)(qrow + 48 + half * 8));

  float m_run = -__builtin_inff();
  float l_run = 0.0f;
  v8f o0 = {}, o1 = {}, o2 = {}, o3 = {};

  __syncthreads();   // bias table ready

  for (int kt = 0; kt < SEQ; kt += 32) {
    // stage K (f16, vector copy) and V (f32 x -> f16, transposed)
    #pragma unroll
    for (int i = 0; i < 2; ++i) {
      const int g  = tid + i * 256;          // 0..511
      const int r  = g >> 4;                 // key 0..31
      const int c4 = (g & 15) << 2;          // hd  0..60
      const size_t gb = ((size_t)(b * SEQ + kt + r)) * D_MODEL + h * HD + c4;
      *(v4h*)&Ks[r][c4] = *(const v4h*)(Kbuf + gb);
      v4h vv = __builtin_convertvector(*(const v4f*)(X + gb), v4h);
      VsT[c4 + 0][r] = vv[0];
      VsT[c4 + 1][r] = vv[1];
      VsT[c4 + 2][r] = vv[2];
      VsT[c4 + 3][r] = vv[3];
    }
    __syncthreads();

    // --- S tiles (keys kt..kt+15 and kt+16..kt+31), hd over 2 wmmas each ---
    v8f s0 = {}, s1 = {};
    {
      v16h a;
      a = cat8(*(const v8h*)&Ks[nn][half * 8],
               *(const v8h*)&Ks[nn][16 + half * 8]);
      s0 = wmma_f16(a, qf0, s0);
      a = cat8(*(const v8h*)&Ks[nn][32 + half * 8],
               *(const v8h*)&Ks[nn][48 + half * 8]);
      s0 = wmma_f16(a, qf1, s0);
      a = cat8(*(const v8h*)&Ks[16 + nn][half * 8],
               *(const v8h*)&Ks[16 + nn][16 + half * 8]);
      s1 = wmma_f16(a, qf0, s1);
      a = cat8(*(const v8h*)&Ks[16 + nn][32 + half * 8],
               *(const v8h*)&Ks[16 + nn][48 + half * 8]);
      s1 = wmma_f16(a, qf1, s1);
    }

    // --- relative-position bias + online softmax ---
    float p0[8], p1[8];
    float mt = -__builtin_inff();
    #pragma unroll
    for (int r = 0; r < 8; ++r) {
      const int k0g = kt + r + half * 8;   // tile0 key (C layout M = r+half*8)
      const int k1g = k0g + 16;
      int rh = clampi(hq - (k0g >> 5) + MAXDIST, 0, 2 * MAXDIST);
      int rw = clampi(wq - (k0g & 31) + MAXDIST, 0, 2 * MAXDIST);
      float v0 = s0[r] + bt[(rh * (2 * MAXDIST + 1) + rw) * NHEADS + h];
      rh = clampi(hq - (k1g >> 5) + MAXDIST, 0, 2 * MAXDIST);
      rw = clampi(wq - (k1g & 31) + MAXDIST, 0, 2 * MAXDIST);
      float v1 = s1[r] + bt[(rh * (2 * MAXDIST + 1) + rw) * NHEADS + h];
      p0[r] = v0; p1[r] = v1;
      mt = fmaxf(mt, fmaxf(v0, v1));
    }
    mt = fmaxf(mt, __shfl_xor(mt, 16, 32));
    const float m_new = fmaxf(m_run, mt);
    const float alpha = __expf(m_run - m_new);
    float lsum = 0.0f;
    #pragma unroll
    for (int r = 0; r < 8; ++r) {
      p0[r] = __expf(p0[r] - m_new);
      p1[r] = __expf(p1[r] - m_new);
      lsum += p0[r] + p1[r];
    }
    lsum += __shfl_xor(lsum, 16, 32);
    l_run = l_run * alpha + lsum;
    m_run = m_new;
    #pragma unroll
    for (int r = 0; r < 8; ++r) {
      o0[r] *= alpha; o1[r] *= alpha; o2[r] *= alpha; o3[r] *= alpha;
    }

    // --- re-lay P (two C-layout tiles) into one 32-key B fragment ---
    v16h pb;
    #pragma unroll
    for (int e = 0; e < 8; ++e) {
      pb[e]     = (_Float16)p0[e];
      pb[e + 8] = (_Float16)p1[e];
    }

    // --- O += V^T(A) x P(B), 4 hd-chunks of 16 ---
    {
      v16h av;
      av = cat8(*(const v8h*)&VsT[nn][half * 8],
                *(const v8h*)&VsT[nn][16 + half * 8]);
      o0 = wmma_f16(av, pb, o0);
      av = cat8(*(const v8h*)&VsT[16 + nn][half * 8],
                *(const v8h*)&VsT[16 + nn][16 + half * 8]);
      o1 = wmma_f16(av, pb, o1);
      av = cat8(*(const v8h*)&VsT[32 + nn][half * 8],
                *(const v8h*)&VsT[32 + nn][16 + half * 8]);
      o2 = wmma_f16(av, pb, o2);
      av = cat8(*(const v8h*)&VsT[48 + nn][half * 8],
                *(const v8h*)&VsT[48 + nn][16 + half * 8]);
      o3 = wmma_f16(av, pb, o3);
    }
    __syncthreads();
  }

  // normalize + vectorized f16 output (4x b128 stores)
  const float inv_l = 1.0f / l_run;
  _Float16* orow = Obuf + ((size_t)(b * SEQ + qg)) * D_MODEL + h * HD;
  v8h h8;
  #pragma unroll
  for (int r = 0; r < 8; ++r) h8[r] = (_Float16)(o0[r] * inv_l);
  *(v8h*)(orow + half * 8) = h8;
  #pragma unroll
  for (int r = 0; r < 8; ++r) h8[r] = (_Float16)(o1[r] * inv_l);
  *(v8h*)(orow + 16 + half * 8) = h8;
  #pragma unroll
  for (int r = 0; r < 8; ++r) h8[r] = (_Float16)(o2[r] * inv_l);
  *(v8h*)(orow + 32 + half * 8) = h8;
  #pragma unroll
  for (int r = 0; r < 8; ++r) h8[r] = (_Float16)(o3[r] * inv_l);
  *(v8h*)(orow + 48 + half * 8) = h8;
}

// ---------------------------------------------------------------------------
// Kernel 3: final = x + (1-g)*ape + g*(O @ Wo^T + bo),  g = sigmoid(gate)
// ---------------------------------------------------------------------------
__global__ __launch_bounds__(256)
void out_proj_blend_kernel(const _Float16* __restrict__ Obuf,
                           const float* __restrict__ Wo, const float* __restrict__ bo,
                           const float* __restrict__ X,  const float* __restrict__ ape,
                           const float* __restrict__ gate,
                           float* __restrict__ out)
{
  __shared__ _Float16 As [64][40];
  __shared__ _Float16 BsT[64][40];

  const int m0   = blockIdx.x * 64;
  const int n0   = blockIdx.y * 64;
  const int tid  = threadIdx.x;
  const int lane = tid & 31;
  const int wid  = tid >> 5;
  const int wm   = wid & 3;
  const int wn   = wid >> 2;
  const int nn   = lane & 15;
  const int half = lane >> 4;

  v8f acc0 = {}; v8f acc1 = {};

  for (int kk = 0; kk < D_MODEL; kk += 32) {
    #pragma unroll
    for (int i = 0; i < 2; ++i) {
      const int g = tid + i * 256;
      const int r  = g >> 3;
      const int k4 = (g & 7) << 2;
      *(v4h*)&As[r][k4] = *(const v4h*)(Obuf + (size_t)(m0 + r) * D_MODEL + kk + k4);
      v4f fb = *(const v4f*)(Wo + (size_t)(n0 + r) * D_MODEL + kk + k4);
      *(v4h*)&BsT[r][k4] = __builtin_convertvector(fb, v4h);
    }
    if (kk + 32 < D_MODEL) {
      __builtin_prefetch(Obuf + (size_t)(m0 + (tid & 63)) * D_MODEL + kk + 32, 0, 1);
    }
    __syncthreads();

    const int arow = wm * 16 + nn;
    v16h a  = cat8(*(const v8h*)&As[arow][half * 16],
                   *(const v8h*)&As[arow][half * 16 + 8]);
    const int bcol = wn * 32 + nn;
    v16h b0 = cat8(*(const v8h*)&BsT[bcol][half * 16],
                   *(const v8h*)&BsT[bcol][half * 16 + 8]);
    v16h b1 = cat8(*(const v8h*)&BsT[bcol + 16][half * 16],
                   *(const v8h*)&BsT[bcol + 16][half * 16 + 8]);
    acc0 = wmma_f16(a, b0, acc0);
    acc1 = wmma_f16(a, b1, acc1);
    __syncthreads();
  }

  const float g  = 1.0f / (1.0f + __expf(-gate[0]));
  const float og = 1.0f - g;
  #pragma unroll
  for (int r = 0; r < 8; ++r) {
    const int row = m0 + wm * 16 + r + half * 8;
    const int c0  = n0 + wn * 32 + nn;
    const size_t i0 = (size_t)row * D_MODEL + c0;
    out[i0]      = X[i0]      + og * ape[i0]      + g * (acc0[r] + bo[c0]);
    out[i0 + 16] = X[i0 + 16] + og * ape[i0 + 16] + g * (acc1[r] + bo[c0 + 16]);
  }
}

// ---------------------------------------------------------------------------
extern "C" void kernel_launch(void* const* d_in, const int* in_sizes, int n_in,
                              void* d_out, int out_size, void* d_ws, size_t ws_size,
                              hipStream_t stream)
{
  const float* x    = (const float*)d_in[0];
  const float* ape  = (const float*)d_in[1];
  const float* Wq   = (const float*)d_in[2];
  const float* bq   = (const float*)d_in[3];
  const float* Wk   = (const float*)d_in[4];
  const float* bk   = (const float*)d_in[5];
  const float* Wo   = (const float*)d_in[6];
  const float* bo   = (const float*)d_in[7];
  const float* btab = (const float*)d_in[8];
  const float* gate = (const float*)d_in[9];
  float* out = (float*)d_out;

  const size_t mat = (size_t)BATCH * SEQ * D_MODEL;   // 2M elements
  _Float16* Qbuf = (_Float16*)d_ws;                   // 4 MB
  _Float16* Kbuf = Qbuf + mat;                        // 4 MB
  _Float16* Obuf = Kbuf + mat;                        // 4 MB (12 MB ws total)

  dim3 blk(256);
  gemm_proj_kernel<<<dim3(128, 4, 2), blk, 0, stream>>>(x, Wq, bq, Wk, bk, Qbuf, Kbuf);
  attention_kernel<<<dim3(SEQ / 128, NHEADS, BATCH), blk, 0, stream>>>(Qbuf, Kbuf, x, btab, Obuf);
  out_proj_blend_kernel<<<dim3(128, 4, 1), blk, 0, stream>>>(Obuf, Wo, bo, x, ape, gate, out);
}
